// CausalSelfAttention_24137716204179
// MI455X (gfx1250) — matrix operand order repaired and checked by reading, not compiled, not run on hardware
//
#include <hip/hip_runtime.h>
#include <math.h>

// ---------------------------------------------------------------------------
// CDNA5 (gfx1250) causal self-attention, bf16 WMMA path + TDM tile staging.
//   A (16x32 bf16): lane L holds row m=L&15; half=L>>4; elem e:
//                   K = half*16+e  -> contiguous (k order chosen here, not the ISA table order)
//   B (32x16 bf16): lane L holds col n=L&15; half=L>>4; elem e: K = half*16+e (same k order as A)
//                   -> one contiguous 32B read (perfect for W^T / K^T rows)
//   C/D (16x16 f32): lane L holds col n=L&15; elem v -> row m = half*8+v
// Shared B-tiles are staged into LDS by the Tensor Data Mover
// (tensor_load_to_lds, TENSORcnt) issued by wave 0 of each block.
// ---------------------------------------------------------------------------

typedef __bf16 bf16_t;
typedef __attribute__((ext_vector_type(16))) __bf16 v16bf;
typedef __attribute__((ext_vector_type(8)))  __bf16 v8bf;
typedef __attribute__((ext_vector_type(8)))  float  v8f;
typedef __attribute__((ext_vector_type(4)))  unsigned int v4u;
typedef __attribute__((ext_vector_type(8)))  int v8i;
typedef __attribute__((ext_vector_type(4)))  int v4i;

static constexpr int BATCH = 4;
static constexpr int SEQ   = 2048;
static constexpr int EMB   = 1024;
static constexpr int NHEAD = 16;
static constexpr int HDIM  = 64;
static constexpr int ROWS  = BATCH * SEQ;   // 8192

__device__ __forceinline__ v8f wmma_bf16(v16bf a, v16bf b, v8f c) {
  return __builtin_amdgcn_wmma_f32_16x16x32_bf16(
      /*neg_a=*/false, a, /*neg_b=*/false, b,
      /*c_mod=*/(short)0, c, /*reuse_a=*/false, /*reuse_b=*/false);
}

__device__ __forceinline__ v16bf load_afrag(const bf16_t* p) {
  v8bf lo = *(const v8bf*)p;
  v8bf hi = *(const v8bf*)(p + 8);
  v16bf a;
#pragma unroll
  for (int i = 0; i < 8; ++i) { a[i] = lo[i]; a[8 + i] = hi[i]; }
  return a;
}

// ---------------------------------------------------------------------------
// TDM: async-copy a 2D row-major tile (tile_w x tile_h bf16 elems, row pitch
// `stride` elems) from global into LDS at `lptr`. D# per ISA ch.8:
//  group0: count=1 | lds_addr | global_addr[56:0] | type=2
//  group1: data_size=1(2B) | tensor_dim0/1 = tile dims (no OOB) |
//          tile_dim0/1 | tensor_dim0_stride
// Issued once per block by wave 0; completion via s_wait_tensorcnt.
// This toolchain's builtin takes 6 args: (g0, g1, g2, g3, g_extra, cpol).
// ---------------------------------------------------------------------------
__device__ __forceinline__ void tdm_load_2d(const bf16_t* gptr, bf16_t* lptr,
                                            unsigned tile_w, unsigned tile_h,
                                            unsigned stride) {
  unsigned long long ga = (unsigned long long)(uintptr_t)gptr;
  unsigned lo = (unsigned)(uintptr_t)lptr;   // generic addr[31:0] == LDS offset
  v4u g0;
  g0[0] = 1u;                                         // count=1, user mode
  g0[1] = lo;                                         // lds_addr
  g0[2] = (unsigned)ga;                               // global_addr[31:0]
  g0[3] = (unsigned)((ga >> 32) & 0x1FFFFFFull) | (2u << 30);  // [56:32]|type=2
  v8i g1;
  g1[0] = (int)(1u << 16);                            // data_size=1 (2 bytes)
  g1[1] = (int)(tile_w << 16);                        // tensor_dim0 lo
  g1[2] = (int)((tile_w >> 16) | (tile_h << 16));     // dim0 hi | tensor_dim1 lo
  g1[3] = (int)((tile_h >> 16) | (tile_w << 16));     // dim1 hi | tile_dim0
  g1[4] = (int)(tile_h & 0xFFFFu);                    // tile_dim1 (tile_dim2=0)
  g1[5] = (int)stride;                                // tensor_dim0_stride lo
  g1[6] = 0;                                          // stride hi | dim1_stride
  g1[7] = 0;
  v4i z4 = {0, 0, 0, 0};
  v8i z8 = {0, 0, 0, 0, 0, 0, 0, 0};
  __builtin_amdgcn_tensor_load_to_lds(g0, g1, z4, z4, z8, 0);
}

// ---------------------------------------------------------------------------
// fp32 -> bf16 cast (4 elements/thread)
// ---------------------------------------------------------------------------
__global__ __launch_bounds__(256) void cvt_f32_bf16(
    const float* __restrict__ src, bf16_t* __restrict__ dst, int n) {
  int i = (blockIdx.x * 256 + threadIdx.x) * 4;
  if (i < n) {
    float4 f = *(const float4*)(src + i);
    dst[i + 0] = (bf16_t)f.x;
    dst[i + 1] = (bf16_t)f.y;
    dst[i + 2] = (bf16_t)f.z;
    dst[i + 3] = (bf16_t)f.w;
  }
}

// ---------------------------------------------------------------------------
// Y[M,N] = X[M,K] @ W[N,K]^T + bias[N]   (bf16 in, f32 accum, OutT out)
// block = 256 threads = 8 waves; block tile 128x64; wave tile 16x64.
// W tile (64 rows x 128 k, 16KB) staged in LDS by TDM, shared by all 8 waves.
// ---------------------------------------------------------------------------
template <typename OutT>
__global__ __launch_bounds__(256) void gemm_xwT(
    const bf16_t* __restrict__ X, const bf16_t* __restrict__ W,
    const float* __restrict__ bias, OutT* __restrict__ Y,
    int M, int N, int K) {
  const int lane = threadIdx.x & 31;
  const int wave = threadIdx.x >> 5;
  const int half = lane >> 4;
  const int nl   = lane & 15;
  const int row0 = blockIdx.x * 128 + wave * 16;
  const int col0 = blockIdx.y * 64;

  __shared__ __align__(32) bf16_t Ws[64 * 128];   // [n_local][k_local]

  const bf16_t* xrow = X + (long)(row0 + nl) * K;
  v8f acc[4] = {};
  for (int k0 = 0; k0 < K; k0 += 128) {
    if (wave == 0) {
      tdm_load_2d(W + (long)col0 * K + k0, Ws, 128, 64, (unsigned)K);
      __builtin_amdgcn_s_wait_tensorcnt(0);
    }
    __builtin_prefetch(xrow + k0 + 128, 0, 0);   // global_prefetch_b8
    __syncthreads();
#pragma unroll
    for (int kk = 0; kk < 4; ++kk) {
      v16bf a = load_afrag(xrow + k0 + kk * 32 + half * 16);
#pragma unroll
      for (int nt = 0; nt < 4; ++nt) {
        v16bf b = *(const v16bf*)&Ws[(nt * 16 + nl) * 128 + kk * 32 + half * 16];
        acc[nt] = wmma_bf16(a, b, acc[nt]);
      }
    }
    __syncthreads();
  }
#pragma unroll
  for (int nt = 0; nt < 4; ++nt) {
    int col = col0 + nt * 16 + nl;
    float bv = bias[col];
#pragma unroll
    for (int v = 0; v < 8; ++v) {
      int row = row0 + half * 8 + v;
      Y[(long)row * N + col] = (OutT)(acc[nt][v] + bv);
    }
  }
}

// ---------------------------------------------------------------------------
// Flash attention. grid = (T/128, NHEAD, BATCH), block = 256 (8 waves).
// K tile (32 keys x 64 d) staged via TDM; V tile transposed manually.
// ---------------------------------------------------------------------------
__global__ __launch_bounds__(256) void attn_kernel(
    const bf16_t* __restrict__ Qb, const bf16_t* __restrict__ Kb,
    const bf16_t* __restrict__ Vb, bf16_t* __restrict__ Ob) {
  const int qblk = blockIdx.x;
  const int h    = blockIdx.y;
  const int b    = blockIdx.z;
  const int lane = threadIdx.x & 31;
  const int wave = threadIdx.x >> 5;
  const int half = lane >> 4;
  const int nl   = lane & 15;

  const int rowBase = b * SEQ;
  const int colBase = h * HDIM;
  const int q0 = qblk * 128;
  const int qw = q0 + wave * 16;

  __shared__ __align__(32) bf16_t Ks[32 * 64];       // [key][d]   (TDM)
  __shared__ __align__(32) bf16_t Vs[64 * 32];       // [d][key]   (transposed)
  __shared__ __align__(32) bf16_t Ps[8][16 * 32];    // per-wave P tile

  v16bf aQ[2];
#pragma unroll
  for (int kk = 0; kk < 2; ++kk)
    aQ[kk] = load_afrag(Qb + (long)(rowBase + qw + nl) * EMB + colBase +
                        kk * 32 + half * 16);

  float mrow[8], lrow[8];
  v8f accO[4] = {};
#pragma unroll
  for (int v = 0; v < 8; ++v) { mrow[v] = -__builtin_inff(); lrow[v] = 0.f; }
  const float scale = 0.125f;  // 1/sqrt(64)

  const int ktEnd = q0 + 128;
  for (int kt = 0; kt < ktEnd; kt += 32) {
    // ---- K tile via Tensor Data Mover; V tile transposed by hand ----
    if (wave == 0)
      tdm_load_2d(Kb + (long)(rowBase + kt) * EMB + colBase, Ks, 64, 32, EMB);
    {
      int r = threadIdx.x >> 3;        // 0..31 (key)
      int c = (threadIdx.x & 7) * 8;   // 0..56 (d)
      v8bf vv = *(const v8bf*)(Vb + (long)(rowBase + kt + r) * EMB + colBase + c);
#pragma unroll
      for (int i = 0; i < 8; ++i) Vs[(c + i) * 32 + r] = vv[i];
    }
    if (wave == 0) __builtin_amdgcn_s_wait_tensorcnt(0);
    __syncthreads();

    if (kt <= qw + 15) {  // wave-uniform causal skip
      // ---- S = Q @ K^T ----
      v8f s[2] = {};
#pragma unroll
      for (int nt = 0; nt < 2; ++nt)
#pragma unroll
        for (int kk = 0; kk < 2; ++kk) {
          v16bf bk = *(const v16bf*)&Ks[(nt * 16 + nl) * 64 + kk * 32 + half * 16];
          s[nt] = wmma_bf16(aQ[kk], bk, s[nt]);
        }
      // ---- causal mask + scale ----
#pragma unroll
      for (int nt = 0; nt < 2; ++nt) {
        int key = kt + nt * 16 + nl;
#pragma unroll
        for (int v = 0; v < 8; ++v) {
          int qr = qw + half * 8 + v;
          s[nt][v] = (key <= qr) ? s[nt][v] * scale : -__builtin_inff();
        }
      }
      // ---- online softmax; rows live in 16-lane groups ----
      float rmax[8], rsum[8], p0[8], p1[8];
#pragma unroll
      for (int v = 0; v < 8; ++v) {
        rmax[v] = fmaxf(s[0][v], s[1][v]);
#pragma unroll
        for (int off = 1; off < 16; off <<= 1)
          rmax[v] = fmaxf(rmax[v], __shfl_xor(rmax[v], off, 32));
      }
#pragma unroll
      for (int v = 0; v < 8; ++v) {
        float mnew  = fmaxf(mrow[v], rmax[v]);
        float alpha = __expf(mrow[v] - mnew);
        p0[v] = __expf(s[0][v] - mnew);
        p1[v] = __expf(s[1][v] - mnew);
        mrow[v] = mnew;
        lrow[v] = lrow[v] * alpha;
        rsum[v] = p0[v] + p1[v];
#pragma unroll
        for (int dt = 0; dt < 4; ++dt) accO[dt][v] = accO[dt][v] * alpha;
      }
#pragma unroll
      for (int v = 0; v < 8; ++v) {
#pragma unroll
        for (int off = 1; off < 16; off <<= 1)
          rsum[v] += __shfl_xor(rsum[v], off, 32);
        lrow[v] += rsum[v];
      }
      // ---- C-layout P -> LDS -> A-layout frag (same-wave DS in-order) ----
      bf16_t* pw = Ps[wave];
#pragma unroll
      for (int v = 0; v < 8; ++v) {
        int r = half * 8 + v;
        pw[r * 32 + nl]      = (bf16_t)p0[v];
        pw[r * 32 + 16 + nl] = (bf16_t)p1[v];
      }
      v16bf aP = load_afrag(pw + nl * 32 + half * 16);
      // ---- O += P @ V ----
#pragma unroll
      for (int dt = 0; dt < 4; ++dt) {
        v16bf bv = *(const v16bf*)&Vs[(dt * 16 + nl) * 32 + half * 16];
        accO[dt] = wmma_bf16(aP, bv, accO[dt]);
      }
    }
    __syncthreads();
  }

#pragma unroll
  for (int v = 0; v < 8; ++v) {
    float inv = 1.0f / lrow[v];
    long row = (long)(rowBase + qw + half * 8 + v);
#pragma unroll
    for (int dt = 0; dt < 4; ++dt)
      Ob[row * EMB + colBase + dt * 16 + nl] = (bf16_t)(accO[dt][v] * inv);
  }
}

// ---------------------------------------------------------------------------
extern "C" void kernel_launch(void* const* d_in, const int* in_sizes, int n_in,
                              void* d_out, int out_size, void* d_ws, size_t ws_size,
                              hipStream_t stream) {
  const float* x  = (const float*)d_in[0];
  const float* Wq = (const float*)d_in[1];
  const float* bq = (const float*)d_in[2];
  const float* Wk = (const float*)d_in[3];
  const float* bk = (const float*)d_in[4];
  const float* Wv = (const float*)d_in[5];
  const float* bv = (const float*)d_in[6];
  const float* Wo = (const float*)d_in[7];
  const float* bo = (const float*)d_in[8];
  float* out = (float*)d_out;

  const long NX = (long)ROWS * EMB;
  const long NW = (long)EMB * EMB;
  bf16_t* xb  = (bf16_t*)d_ws;
  bf16_t* wqb = xb  + NX;
  bf16_t* wkb = wqb + NW;
  bf16_t* wvb = wkb + NW;
  bf16_t* wob = wvb + NW;
  bf16_t* Qb  = wob + NW;
  bf16_t* Kb  = Qb  + NX;
  bf16_t* Vb  = Kb  + NX;
  bf16_t* Ab  = Vb  + NX;   // total ~88 MB

  cvt_f32_bf16<<<(int)(NX / 4 / 256), 256, 0, stream>>>(x,  xb,  (int)NX);
  cvt_f32_bf16<<<(int)(NW / 4 / 256), 256, 0, stream>>>(Wq, wqb, (int)NW);
  cvt_f32_bf16<<<(int)(NW / 4 / 256), 256, 0, stream>>>(Wk, wkb, (int)NW);
  cvt_f32_bf16<<<(int)(NW / 4 / 256), 256, 0, stream>>>(Wv, wvb, (int)NW);
  cvt_f32_bf16<<<(int)(NW / 4 / 256), 256, 0, stream>>>(Wo, wob, (int)NW);

  dim3 gg(ROWS / 128, EMB / 64, 1), bb(256);
  gemm_xwT<bf16_t><<<gg, bb, 0, stream>>>(xb, wqb, bq, Qb, ROWS, EMB, EMB);
  gemm_xwT<bf16_t><<<gg, bb, 0, stream>>>(xb, wkb, bk, Kb, ROWS, EMB, EMB);
  gemm_xwT<bf16_t><<<gg, bb, 0, stream>>>(xb, wvb, bv, Vb, ROWS, EMB, EMB);

  attn_kernel<<<dim3(SEQ / 128, NHEAD, BATCH), 256, 0, stream>>>(Qb, Kb, Vb, Ab);

  gemm_xwT<float><<<gg, bb, 0, stream>>>(Ab, wob, bo, out, ROWS, EMB, EMB);
}
